// MultiHeadSelfAttentionLayer_24747601559689
// MI455X (gfx1250) — matrix operand order repaired and checked by reading, not compiled, not run on hardware
//
#include <hip/hip_runtime.h>
#include <hip/hip_bf16.h>

typedef __attribute__((ext_vector_type(16))) __bf16 v16bf;
typedef __attribute__((ext_vector_type(8)))  __bf16 v8bf;
typedef __attribute__((ext_vector_type(8)))  float  v8f;

#define N_TOK   2048
#define D_MODEL 1024
#define N_HEADS 16
#define D_HEAD  64
#define BATCH   4
#define M_ROWS  (BATCH * N_TOK)   // 8192

static __device__ __forceinline__ v8f wmma_bf16(v16bf a, v16bf b, v8f c) {
    return __builtin_amdgcn_wmma_f32_16x16x32_bf16(false, a, false, b, (short)0, c, false, false);
}

// ---- 16-lane butterfly reductions, pure DPP (no LDS round trips) ----------
template<int CTRL>
static __device__ __forceinline__ float dppf(float x) {
    int t = __builtin_amdgcn_update_dpp(0, __builtin_bit_cast(int, x),
                                        CTRL, 0xF, 0xF, true);
    return __builtin_bit_cast(float, t);
}
static __device__ __forceinline__ float red_max16(float x) {
    x = fmaxf(x, dppf<0xB1>(x));   // quad_perm [1,0,3,2]  (xor 1)
    x = fmaxf(x, dppf<0x4E>(x));   // quad_perm [2,3,0,1]  (xor 2)
    x = fmaxf(x, dppf<0x141>(x));  // row_half_mirror      (xor 4 eff.)
    x = fmaxf(x, dppf<0x140>(x));  // row_mirror           (xor 8 eff.)
    return x;
}
static __device__ __forceinline__ float red_add16(float x) {
    x += dppf<0xB1>(x);
    x += dppf<0x4E>(x);
    x += dppf<0x141>(x);
    x += dppf<0x140>(x);
    return x;
}

// ---------------------------------------------------------------- LayerNorm
__global__ __launch_bounds__(256)
void ln_kernel(const float* __restrict__ z, const float* __restrict__ gamma,
               const float* __restrict__ beta, __bf16* __restrict__ zn) {
    int row = blockIdx.x;
    const float* zr = z + (size_t)row * D_MODEL;
    float s = 0.f, s2 = 0.f;
    #pragma unroll
    for (int c0 = 0; c0 < D_MODEL; c0 += 256) {
        float x = zr[c0 + threadIdx.x]; s += x; s2 += x * x;
    }
    s  = red_add16(s);
    s2 = red_add16(s2);
    // fold the two 16-lane halves of the wave, then across 8 waves via LDS
    s  += __shfl_xor(s, 16, 32);
    s2 += __shfl_xor(s2, 16, 32);
    __shared__ float red0[8], red1[8];
    int wv = threadIdx.x >> 5, ln = threadIdx.x & 31;
    if (ln == 0) { red0[wv] = s; red1[wv] = s2; }
    __syncthreads();
    float ts = 0.f, ts2 = 0.f;
    #pragma unroll
    for (int i = 0; i < 8; i++) { ts += red0[i]; ts2 += red1[i]; }
    float mu  = ts * (1.f / D_MODEL);
    float var = ts2 * (1.f / D_MODEL) - mu * mu;
    float inv = rsqrtf(var + 1e-5f);
    #pragma unroll
    for (int c0 = 0; c0 < D_MODEL; c0 += 256) {
        int c = c0 + threadIdx.x;
        float x = (zr[c] - mu) * inv * gamma[c] + beta[c];
        zn[(size_t)row * D_MODEL + c] = (__bf16)x;
    }
}

// ------------------------------------------- weight fp32 -> bf16 transposed
__global__ __launch_bounds__(256)
void cvt_w_kernel(const float* __restrict__ wqkv, const float* __restrict__ wproj,
                  __bf16* __restrict__ wqkvt, __bf16* __restrict__ wprojt) {
    const int TOT1 = 3 * D_MODEL * D_MODEL;
    const int TOT2 = D_MODEL * D_MODEL;
    int idx = blockIdx.x * 256 + threadIdx.x;
    if (idx < TOT1) {
        int nt = idx / D_MODEL, k = idx % D_MODEL;
        wqkvt[idx] = (__bf16)wqkv[(size_t)k * (3 * D_MODEL) + nt];
    } else {
        int i2 = idx - TOT1;
        if (i2 < TOT2) {
            int nt = i2 / D_MODEL, k = i2 % D_MODEL;
            wprojt[i2] = (__bf16)wproj[(size_t)k * D_MODEL + nt];
        }
    }
}

// ------------------------------------------------------------- GEMM template
// C[M x N] = A[M x K] * Bt[N x K]^T, bf16 in, f32 accum.
// BM=64, BN=128, BK=32; 256 thr = 8 waves (4 M-strips x 2 N-strips).
// Tiles stream global->LDS via async direct-to-LDS, double buffered (ASYNCcnt).
template<int MODE>
__global__ __launch_bounds__(256)
void gemm_kernel(const __bf16* __restrict__ A, const __bf16* __restrict__ Bt,
                 __bf16* __restrict__ Qo, __bf16* __restrict__ Ko, __bf16* __restrict__ Vto,
                 const float* __restrict__ zres, const float* __restrict__ bias,
                 float* __restrict__ out) {
    const int K = D_MODEL;
    int bm = blockIdx.y, bn = blockIdx.x;
    __shared__ __bf16 sA[2][64][32];
    __shared__ __bf16 sB[2][128][32];
    int tid = threadIdx.x, wv = tid >> 5, ln = tid & 31;
    int wm = wv & 3, wn = wv >> 2;
    int r = ln & 15, half = ln >> 4;

    v8f acc[4];
    #pragma unroll
    for (int t = 0; t < 4; t++)
        #pragma unroll
        for (int e = 0; e < 8; e++) acc[t][e] = 0.f;

    int arow = tid >> 2, acg = tid & 3;    // A: 64 rows x (4 groups of 8 elts)
    int brow = tid >> 1, bhf = tid & 1;    // B: 128 rows x (2 halves of 16 elts)

    const __bf16* aSrc = A  + (size_t)(bm * 64  + arow) * K + acg * 8;
    const __bf16* bSrc = Bt + (size_t)(bn * 128 + brow) * K + bhf * 16;
    unsigned aDst = (unsigned)(uintptr_t)&sA[0][arow][acg * 8];
    unsigned bDst = (unsigned)(uintptr_t)&sB[0][brow][bhf * 16];
    const unsigned aStride = 64 * 32 * 2;    // bytes per A buffer
    const unsigned bStride = 128 * 32 * 2;   // bytes per B buffer

#define GEMM_ISSUE(k0, buf)                                                    \
    do {                                                                       \
        asm volatile("global_load_async_to_lds_b128 %0, %1, off"               \
                     :: "v"(aDst + (buf) * aStride), "v"(aSrc + (k0))          \
                     : "memory");                                              \
        asm volatile("global_load_async_to_lds_b128 %0, %1, off"               \
                     :: "v"(bDst + (buf) * bStride), "v"(bSrc + (k0))          \
                     : "memory");                                              \
        asm volatile("global_load_async_to_lds_b128 %0, %1, off"               \
                     :: "v"(bDst + (buf) * bStride + 16), "v"(bSrc + (k0) + 8) \
                     : "memory");                                              \
    } while (0)

    GEMM_ISSUE(0, 0);

    for (int kt = 0; kt < K / 32; ++kt) {
        int buf = kt & 1;
        if (kt < K / 32 - 1) {
            GEMM_ISSUE((kt + 1) * 32, buf ^ 1);
            asm volatile("s_wait_asynccnt 3" ::: "memory");
        } else {
            asm volatile("s_wait_asynccnt 0" ::: "memory");
        }
        __syncthreads();

        v16bf af;
        {
            const __bf16* ap = &sA[buf][wm * 16 + r][0];
            v8bf lo = *(const v8bf*)&ap[half * 16];
            v8bf hi = *(const v8bf*)&ap[half * 16 + 8];
            #pragma unroll
            for (int e = 0; e < 8; e++) { af[e] = lo[e]; af[8 + e] = hi[e]; }
        }
        #pragma unroll
        for (int nt = 0; nt < 4; nt++) {
            v16bf bfr = *(const v16bf*)&sB[buf][wn * 64 + nt * 16 + r][half * 16];
            acc[nt] = wmma_bf16(af, bfr, acc[nt]);
        }
        __syncthreads();
    }
#undef GEMM_ISSUE

    #pragma unroll
    for (int nt = 0; nt < 4; nt++) {
        int col = bn * 128 + wn * 64 + nt * 16 + r;
        #pragma unroll
        for (int i = 0; i < 8; i++) {
            int row = bm * 64 + wm * 16 + i + 8 * half;
            float v = acc[nt][i];
            if (MODE == 0) {
                int which = col % 3;
                int hh = col / (3 * D_HEAD);
                int di = (col % (3 * D_HEAD)) / 3;
                int bb = row >> 11;
                int q  = row & (N_TOK - 1);
                size_t bh = (size_t)bb * N_HEADS + hh;
                if (which == 0)
                    Qo[(bh * N_TOK + q) * D_HEAD + di] = (__bf16)v;
                else if (which == 1)
                    Ko[(bh * N_TOK + q) * D_HEAD + di] = (__bf16)v;
                else
                    Vto[(bh * D_HEAD + di) * N_TOK + q] = (__bf16)v;
            } else {
                size_t oi = (size_t)row * D_MODEL + col;
                out[oi] = v + bias[col] + zres[oi];
            }
        }
    }
}

// --------------------------------------------------------- flash attention
// grid (16, 64): x = 128-query-row block, y = (b*16 + h)
__global__ __launch_bounds__(256)
void attn_kernel(const __bf16* __restrict__ Q, const __bf16* __restrict__ Kv,
                 const __bf16* __restrict__ Vt, __bf16* __restrict__ attn_out) {
    int bh = blockIdx.y;
    int qb = blockIdx.x;
    int tid = threadIdx.x, wv = tid >> 5, ln = tid & 31;
    int r = ln & 15, half = ln >> 4;
    int qbase = qb * 128 + wv * 16;

    const __bf16* Qb = Q  + (size_t)bh * N_TOK * D_HEAD;
    const __bf16* Kb = Kv + (size_t)bh * N_TOK * D_HEAD;
    const __bf16* Vb = Vt + (size_t)bh * D_HEAD * N_TOK;

    // resident Q fragments (K-dim 64 -> 2 WMMA k-steps)
    v16bf aq[2];
    #pragma unroll
    for (int ks = 0; ks < 2; ks++) {
        const __bf16* qp = Qb + (size_t)(qbase + r) * D_HEAD + ks * 32;
        v8bf lo = *(const v8bf*)&qp[half * 16];
        v8bf hi = *(const v8bf*)&qp[half * 16 + 8];
        #pragma unroll
        for (int e = 0; e < 8; e++) { aq[ks][e] = lo[e]; aq[ks][8 + e] = hi[e]; }
    }

    float m_i[8], l_i[8];
    v8f acc[4];
    #pragma unroll
    for (int i = 0; i < 8; i++) { m_i[i] = -1e30f; l_i[i] = 0.f; }
    #pragma unroll
    for (int t = 0; t < 4; t++)
        #pragma unroll
        for (int e = 0; e < 8; e++) acc[t][e] = 0.f;

    __shared__ __bf16 sP[8][16][32];   // per-wave P tile (C-layout -> A-layout)

    for (int kc = 0; kc < N_TOK; kc += 32) {
        // batch-issue all K and V fragment loads for this chunk
        v16bf bk[2][2], bv[4];
        #pragma unroll
        for (int nt = 0; nt < 2; nt++)
            #pragma unroll
            for (int ks = 0; ks < 2; ks++)
                bk[nt][ks] = *(const v16bf*)(Kb +
                    (size_t)(kc + nt * 16 + r) * D_HEAD + ks * 32 + half * 16);
        #pragma unroll
        for (int nt = 0; nt < 4; nt++)
            bv[nt] = *(const v16bf*)(Vb +
                (size_t)(nt * 16 + r) * N_TOK + kc + half * 16);

        // S = Q * K^T  (16 x 32 as two 16x16 tiles)
        v8f s[2];
        #pragma unroll
        for (int t = 0; t < 2; t++)
            #pragma unroll
            for (int e = 0; e < 8; e++) s[t][e] = 0.f;
        #pragma unroll
        for (int nt = 0; nt < 2; nt++)
            #pragma unroll
            for (int ks = 0; ks < 2; ks++)
                s[nt] = wmma_bf16(aq[ks], bk[nt][ks], s[nt]);

        // online softmax; rows map M = i + 8*half, stats per 16-lane group
        const float sc = 0.125f;               // 1/sqrt(64)
        #pragma unroll
        for (int i = 0; i < 8; i++) {
            float v0 = s[0][i] * sc, v1 = s[1][i] * sc;
            float mx = red_max16(fmaxf(v0, v1));
            float mnew = fmaxf(m_i[i], mx);
            float al = __expf(m_i[i] - mnew);
            float p0 = __expf(v0 - mnew);
            float p1 = __expf(v1 - mnew);
            s[0][i] = p0; s[1][i] = p1;
            float rs = red_add16(p0 + p1);
            l_i[i] = l_i[i] * al + rs;
            m_i[i] = mnew;
            #pragma unroll
            for (int t = 0; t < 4; t++) acc[t][i] *= al;
        }

        // C-layout -> A-layout through per-wave LDS tile
        #pragma unroll
        for (int i = 0; i < 8; i++) {
            sP[wv][i + 8 * half][r]      = (__bf16)s[0][i];
            sP[wv][i + 8 * half][16 + r] = (__bf16)s[1][i];
        }
        __builtin_amdgcn_wave_barrier();
        v16bf ap;
        {
            const __bf16* pp = &sP[wv][r][0];
            v8bf lo = *(const v8bf*)&pp[half * 16];
            v8bf hi = *(const v8bf*)&pp[half * 16 + 8];
            #pragma unroll
            for (int e = 0; e < 8; e++) { ap[e] = lo[e]; ap[8 + e] = hi[e]; }
        }
        __builtin_amdgcn_wave_barrier();

        // acc += P * V   (Vt is [d, n]: contiguous along keys)
        #pragma unroll
        for (int nt = 0; nt < 4; nt++)
            acc[nt] = wmma_bf16(ap, bv[nt], acc[nt]);
    }

    // normalize + store attn_out [M_ROWS, D_MODEL] (bf16)
    int hh = bh & (N_HEADS - 1);
    int bb = bh >> 4;
    #pragma unroll
    for (int i = 0; i < 8; i++) {
        float inv = 1.f / l_i[i];
        int qg = qbase + i + 8 * half;
        size_t rowg = (size_t)bb * N_TOK + qg;
        #pragma unroll
        for (int t = 0; t < 4; t++) {
            int col = hh * D_HEAD + t * 16 + r;
            attn_out[rowg * D_MODEL + col] = (__bf16)(acc[t][i] * inv);
        }
    }
}

// ------------------------------------------------------------------- launch
extern "C" void kernel_launch(void* const* d_in, const int* in_sizes, int n_in,
                              void* d_out, int out_size, void* d_ws, size_t ws_size,
                              hipStream_t stream) {
    const float* z      = (const float*)d_in[0];
    const float* gamma  = (const float*)d_in[1];
    const float* beta   = (const float*)d_in[2];
    const float* w_qkv  = (const float*)d_in[3];
    const float* w_proj = (const float*)d_in[4];
    const float* b_proj = (const float*)d_in[5];
    float* out = (float*)d_out;

    char* ws = (char*)d_ws;
    size_t off = 0;
    __bf16* zn     = (__bf16*)(ws + off); off += (size_t)M_ROWS * D_MODEL * 2;
    __bf16* wqkvt  = (__bf16*)(ws + off); off += (size_t)3 * D_MODEL * D_MODEL * 2;
    __bf16* wprojt = (__bf16*)(ws + off); off += (size_t)D_MODEL * D_MODEL * 2;
    __bf16* Qb     = (__bf16*)(ws + off); off += (size_t)M_ROWS * D_MODEL * 2;
    __bf16* Kb     = (__bf16*)(ws + off); off += (size_t)M_ROWS * D_MODEL * 2;
    __bf16* Vtb    = (__bf16*)(ws + off); off += (size_t)M_ROWS * D_MODEL * 2;
    __bf16* attnb  = (__bf16*)(ws + off); off += (size_t)M_ROWS * D_MODEL * 2;

    ln_kernel<<<M_ROWS, 256, 0, stream>>>(z, gamma, beta, zn);

    int cvt_total = 3 * D_MODEL * D_MODEL + D_MODEL * D_MODEL;
    cvt_w_kernel<<<(cvt_total + 255) / 256, 256, 0, stream>>>(w_qkv, w_proj, wqkvt, wprojt);

    gemm_kernel<0><<<dim3(3 * D_MODEL / 128, M_ROWS / 64), 256, 0, stream>>>(
        zn, wqkvt, Qb, Kb, Vtb, nullptr, nullptr, nullptr);

    attn_kernel<<<dim3(N_TOK / 128, BATCH * N_HEADS), 256, 0, stream>>>(Qb, Kb, Vtb, attnb);

    gemm_kernel<1><<<dim3(D_MODEL / 128, M_ROWS / 64), 256, 0, stream>>>(
        attnb, wprojt, nullptr, nullptr, nullptr, z, b_proj, out);
}
